// Decomposition_TransformerBlock_65764539236388
// MI455X (gfx1250) — matrix operand order repaired and checked by reading, not compiled, not run on hardware
//
#include <hip/hip_runtime.h>
#include <stdint.h>

// ---------------------------------------------------------------------------
// Decomposition Transformer block for MI455X (gfx1250, wave32).
// - All GEMMs on v_wmma_f32_16x16x32_bf16 (bf16 operands, f32 accumulate).
// - Weights pre-transposed so every WMMA B-operand is a contiguous 2x16B load.
// - GEMM A-tiles staged by the Tensor Data Mover (tensor_load_to_lds +
//   s_wait_tensorcnt) when available; attention K/V staged with async copies
//   (global_load_async_to_lds_b128 + s_wait_asynccnt). Double-buffered LDS.
// - GEMM: 64-row (32 for decomp epilogue) x 256-col workgroup tiles ->
//   16 WMMAs per k-step per wave.  Attention: 64-key online-softmax steps,
//   8 WMMAs per step per wave.
// ---------------------------------------------------------------------------

typedef __attribute__((ext_vector_type(16))) __bf16 v16bf;
typedef __attribute__((ext_vector_type(8)))  __bf16 v8bf;
typedef __attribute__((ext_vector_type(8)))  float  v8f;
typedef __attribute__((ext_vector_type(4)))  unsigned int v4u;
typedef __attribute__((ext_vector_type(8)))  int v8i;
typedef __attribute__((ext_vector_type(4)))  int v4i;

#define DEV __device__ __forceinline__

#if __has_builtin(__builtin_amdgcn_tensor_load_to_lds)
#define HAVE_TDM 1
#else
#define HAVE_TDM 0
#endif

static constexpr int B_ = 4, S_ = 2048, E_ = 256, H_ = 8, D_ = 32, FF_ = 1024;
static constexpr int T_ = B_ * S_;      // 8192 tokens
static constexpr int PADW = 12;         // (25-1)/2

DEV int laneid() { return (int)(threadIdx.x & 31u); }

DEV v8f zero8() {
  v8f z;
#pragma unroll
  for (int i = 0; i < 8; ++i) z[i] = 0.f;
  return z;
}

DEV v8f wmma_bf16(v16bf a, v16bf b, v8f c) {
  return __builtin_amdgcn_wmma_f32_16x16x32_bf16(false, a, false, b, (short)0, c,
                                                 false, false);
}

// Async 16B copy global -> LDS (per-lane). LDS offset = low 32 bits of the
// generic pointer (LDS aperture lives entirely in the high 32 bits).
DEV void async_copy_b128(const void* gaddr, void* lds) {
  uint32_t lo = (uint32_t)(uintptr_t)lds;
  asm volatile("global_load_async_to_lds_b128 %0, %1, off"
               :: "v"(lo), "v"(gaddr) : "memory");
}
DEV void wait_async() { asm volatile("s_wait_asynccnt 0" ::: "memory"); }

#if HAVE_TDM
// TDM 2D tile load: tile_h rows x tile_w bf16 elems, row stride in elems.
// D# bit layout per CDNA5 ISA ch.8 (type=2 "image", data_size=1 -> 2 bytes).
DEV void tdm_load_tile(const __bf16* gaddr, void* lds, int tile_w, int tile_h,
                       int row_stride) {
  uint64_t ga = (uint64_t)(uintptr_t)gaddr;
  v4u g0;
  g0[0] = 1u;                                               // count=1, user D#
  g0[1] = (uint32_t)(uintptr_t)lds;                         // lds_addr
  g0[2] = (uint32_t)ga;                                     // global_addr[31:0]
  g0[3] = (uint32_t)((ga >> 32) & 0x1FFFFFFu) | (2u << 30); // addr[56:32], type=2
  v8i g1;
  g1[0] = (int)(1u << 16);                    // wg_mask=0, data_size=1 (2B)
  g1[1] = (int)((uint32_t)tile_w << 16);      // tensor_dim0[15:0]
  g1[2] = (int)((uint32_t)tile_h << 16);      // tensor_dim1[15:0]
  g1[3] = (int)((uint32_t)tile_w << 16);      // tile_dim0
  g1[4] = (int)(uint32_t)tile_h;              // tile_dim1 (tile_dim2=0)
  g1[5] = row_stride;                         // tensor_dim0_stride[31:0]
  g1[6] = 0;                                  // stride hi, dim1_stride lo
  g1[7] = 0;
  v4i z4 = {0, 0, 0, 0};
  v8i z8;
#pragma unroll
  for (int i = 0; i < 8; ++i) z8[i] = 0;
  __builtin_amdgcn_tensor_load_to_lds(g0, g1, z4, z4, z8, 0);
}
DEV void wait_tensor() {
#if __has_builtin(__builtin_amdgcn_s_wait_tensorcnt)
  __builtin_amdgcn_s_wait_tensorcnt(0);
#else
  asm volatile("s_wait_tensorcnt 0" ::: "memory");
#endif
}
#endif

// A fragment 16x32 bf16 from row-major src[m*ld + k]; base/ld 16B aligned.
// Lanes 0-15: row M=lane, K=0..15; lanes 16-31: row M=lane-16, K=16..31.
DEV v16bf ldA(const __bf16* src, int ld) {
  int l = laneid();
  int m = l & 15;
  int kb = (l < 16) ? 0 : 16;
  const __bf16* p = src + (size_t)m * ld + kb;
  union { v16bf v; v8bf h[2]; } u;
  u.h[0] = *(const v8bf*)(p);
  u.h[1] = *(const v8bf*)(p + 8);
  return u.v;
}

// B fragment 32x16 bf16, reduction dim contiguous in memory: src[n*ld + k].
// Lanes 0-15: N=lane, K=0..15; lanes 16-31: N=lane-16, K=16..31.
DEV v16bf ldBt(const __bf16* src, int ld) {
  int l = laneid();
  int n = l & 15;
  int kb = (l < 16) ? 0 : 16;
  const __bf16* p = src + (size_t)n * ld + kb;
  union { v16bf v; v8bf h[2]; } u;
  u.h[0] = *(const v8bf*)(p);
  u.h[1] = *(const v8bf*)(p + 8);
  return u.v;
}

// C fragment store (16x16 f32 -> bf16), row-major dst[m*ld + n], scaled.
DEV void stC_bf16(__bf16* dst, int ld, v8f c, float scale) {
  int l = laneid(); int n = l & 15; int mb = (l < 16) ? 0 : 8;
#pragma unroll
  for (int r = 0; r < 8; ++r) dst[(size_t)(mb + r) * ld + n] = (__bf16)(c[r] * scale);
}

// Transposed C store: dst[n*ldT + m] (builds V^T).
DEV void stC_T_bf16(__bf16* dst, int ldT, v8f c) {
  int l = laneid(); int n = l & 15; int mb = (l < 16) ? 0 : 8;
#pragma unroll
  for (int r = 0; r < 8; ++r) dst[(size_t)n * ldT + mb + r] = (__bf16)c[r];
}

// ---------------------------------------------------------------------------
__global__ void k_cvt_bf16(const float* __restrict__ src, __bf16* __restrict__ dst,
                           int n) {
  int i = blockIdx.x * blockDim.x + threadIdx.x;
  if (i < n) dst[i] = (__bf16)src[i];
}

// Convert + transpose: src is [R][C] f32, dst is [C][R] bf16.
__global__ void k_cvt_bf16_T(const float* __restrict__ src, __bf16* __restrict__ dst,
                             int R, int C) {
  int i = blockIdx.x * blockDim.x + threadIdx.x;
  if (i < R * C) {
    int r = i / C, c = i % C;
    dst[(size_t)c * R + r] = (__bf16)src[i];
  }
}

// ---------------------------------------------------------------------------
// QKV projection: per (b,h) 16-token tile, K = D = 32 (one WMMA reduction).
// q scaled by 1/sqrt(E)=1/16.  V stored transposed: vTT[bh][d][s].
__global__ void k_qkv(const __bf16* __restrict__ xb, const __bf16* __restrict__ wqT,
                      const __bf16* __restrict__ wkT, const __bf16* __restrict__ wvT,
                      __bf16* __restrict__ qT, __bf16* __restrict__ kT,
                      __bf16* __restrict__ vTT) {
  int bh = blockIdx.x;
  int b = bh / H_, h = bh % H_;
  int wave = (int)(threadIdx.x >> 5);
  int t0 = blockIdx.y * 64 + wave * 16;

  v16bf a = ldA(xb + (size_t)(b * S_ + t0) * E_ + h * D_, E_);
  v8f z = zero8();
  v16bf bq0 = ldBt(wqT, D_),           bq1 = ldBt(wqT + 16 * D_, D_);
  v16bf bk0 = ldBt(wkT, D_),           bk1 = ldBt(wkT + 16 * D_, D_);
  v16bf bv0 = ldBt(wvT, D_),           bv1 = ldBt(wvT + 16 * D_, D_);
  v8f cq0 = wmma_bf16(a, bq0, z), cq1 = wmma_bf16(a, bq1, z);
  v8f ck0 = wmma_bf16(a, bk0, z), ck1 = wmma_bf16(a, bk1, z);
  v8f cv0 = wmma_bf16(a, bv0, z), cv1 = wmma_bf16(a, bv1, z);

  __bf16* qd = qT + (size_t)(bh * S_ + t0) * D_;
  __bf16* kd = kT + (size_t)(bh * S_ + t0) * D_;
  stC_bf16(qd,      D_, cq0, 0.0625f);
  stC_bf16(qd + 16, D_, cq1, 0.0625f);
  stC_bf16(kd,      D_, ck0, 1.f);
  stC_bf16(kd + 16, D_, ck1, 1.f);
  __bf16* vd = vTT + (size_t)bh * D_ * S_ + t0;
  stC_T_bf16(vd, S_, cv0);
  stC_T_bf16(vd + (size_t)16 * S_, S_, cv1);
}

// ---------------------------------------------------------------------------
// Flash attention. 4 waves/block share (b,h); each wave owns a 16-query tile.
// 64-key chunks staged once per block via async copies, double-buffered.
// 8 WMMAs + one online-softmax update per chunk.
__global__ void k_attn(const __bf16* __restrict__ qT, const __bf16* __restrict__ kT,
                       const __bf16* __restrict__ vTT, const int* __restrict__ mask,
                       __bf16* __restrict__ aob) {
  __shared__ __align__(16) __bf16 kbuf[2][64 * 32];   // [token][d]
  __shared__ __align__(16) __bf16 vbuf[2][32 * 64];   // [d][token]
  __shared__ __align__(16) __bf16 psh[4][16 * 64];    // per-wave P bounce
  int bh = blockIdx.x;
  int b = bh / H_, h = bh % H_;
  int wave = (int)(threadIdx.x >> 5);
  int q0 = blockIdx.y * 64 + wave * 16;
  int l = laneid();
  int n = l & 15;
  int mb = (l < 16) ? 0 : 8;
  int tid = (int)threadIdx.x;

  const __bf16* kbh = kT + (size_t)bh * S_ * D_;
  const __bf16* vbh = vTT + (size_t)bh * D_ * S_;

  auto stage = [&](int buf, int kc) {
    for (int t = tid; t < 256; t += 128) {
      int krow = t >> 2, kseg = (t & 3) * 8;          // 64 rows x 4 x 8 elems
      async_copy_b128(kbh + (size_t)(kc + krow) * D_ + kseg,
                      &kbuf[buf][krow * 32 + kseg]);
      int drow = t >> 3, vseg = (t & 7) * 8;          // 32 rows x 8 x 8 elems
      async_copy_b128(vbh + (size_t)drow * S_ + kc + vseg,
                      &vbuf[buf][drow * 64 + vseg]);
    }
  };

  v16bf aq = ldA(qT + (size_t)(bh * S_ + q0) * D_, D_);
  v8f o0 = zero8(), o1 = zero8();
  float m8[8], l8[8];
#pragma unroll
  for (int r = 0; r < 8; ++r) { m8[r] = -3.0e38f; l8[r] = 0.f; }
  __bf16* ps = psh[wave];

  stage(0, 0);
  wait_async();
  __syncthreads();
  int cur = 0;

  for (int kc = 0; kc < S_; kc += 64) {
    if (kc + 64 < S_) stage(cur ^ 1, kc + 64);

    // scores = Q . K^T for 64 keys (4 fragments)
    v8f s[4];
#pragma unroll
    for (int f = 0; f < 4; ++f) {
      v16bf bk = ldBt(kbuf[cur] + f * 16 * 32, 32);
      s[f] = wmma_bf16(aq, bk, zero8());
    }
#pragma unroll
    for (int f = 0; f < 4; ++f) {
      int mk = mask[b * S_ + kc + f * 16 + n];
      if (mk == 0) {
#pragma unroll
        for (int r = 0; r < 8; ++r) s[f][r] = -1e20f;
      }
    }

    // per-row chunk max across fragments, then across the 16 owning lanes
    float mx[8];
#pragma unroll
    for (int r = 0; r < 8; ++r)
      mx[r] = fmaxf(fmaxf(s[0][r], s[1][r]), fmaxf(s[2][r], s[3][r]));
#pragma unroll
    for (int off = 1; off <= 8; off <<= 1) {
#pragma unroll
      for (int r = 0; r < 8; ++r) mx[r] = fmaxf(mx[r], __shfl_xor(mx[r], off, 32));
    }

    float rs[8];
#pragma unroll
    for (int r = 0; r < 8; ++r) {
      float mn = fmaxf(m8[r], mx[r]);
      float sc = __expf(m8[r] - mn);
      m8[r] = mn;
#pragma unroll
      for (int f = 0; f < 4; ++f) s[f][r] = __expf(s[f][r] - mn);
      l8[r] *= sc;
      o0[r] *= sc;
      o1[r] *= sc;
      rs[r] = (s[0][r] + s[1][r]) + (s[2][r] + s[3][r]);
    }
#pragma unroll
    for (int off = 1; off <= 8; off <<= 1) {
#pragma unroll
      for (int r = 0; r < 8; ++r) rs[r] += __shfl_xor(rs[r], off, 32);
    }
#pragma unroll
    for (int r = 0; r < 8; ++r) l8[r] += rs[r];

    // C-layout -> A-layout bounce through wave-private LDS (DScnt orders RAW)
#pragma unroll
    for (int f = 0; f < 4; ++f) {
#pragma unroll
      for (int r = 0; r < 8; ++r) ps[(mb + r) * 64 + f * 16 + n] = (__bf16)s[f][r];
    }
    v16bf ap0 = ldA(ps, 64);
    v16bf ap1 = ldA(ps + 32, 64);
    o0 = wmma_bf16(ap0, ldBt(vbuf[cur], 64), o0);
    o0 = wmma_bf16(ap1, ldBt(vbuf[cur] + 32, 64), o0);
    o1 = wmma_bf16(ap0, ldBt(vbuf[cur] + 16 * 64, 64), o1);
    o1 = wmma_bf16(ap1, ldBt(vbuf[cur] + 16 * 64 + 32, 64), o1);

    wait_async();
    __syncthreads();
    cur ^= 1;
  }

  size_t rowbase = (size_t)(b * S_ + q0 + mb) * E_ + h * D_;
#pragma unroll
  for (int r = 0; r < 8; ++r) {
    float inv = 1.f / l8[r];
    aob[rowbase + (size_t)r * E_ + n]      = (__bf16)(o0[r] * inv);
    aob[rowbase + (size_t)r * E_ + 16 + n] = (__bf16)(o1[r] * inv);
  }
}

// ---------------------------------------------------------------------------
// Generic WMMA GEMM. Workgroup tile: MT rows x 256 cols; each of 4 waves owns
// 64 cols across all MT rows -> (MT/16)*4 WMMAs per k-step per wave.
// Wt is pre-transposed [N][K]. The A MTx32 tile is staged to LDS per k-step by
// the TDM (one descriptor DMA, wave 0) or async copies, double-buffered.
// Epilogues: +bias, optional ReLU, optional (+residual, series decomposition
// along the 256 feature columns via sliding-window 25-tap moving average).
// OF/OB select f32/bf16 output at compile time.
template <int MT, bool RELU, bool DECOMP, bool OF, bool OB>
__global__ void k_gemm(const __bf16* __restrict__ A, const __bf16* __restrict__ Wt,
                       const float* __restrict__ bias, const float* __restrict__ resid,
                       float* __restrict__ outF, __bf16* __restrict__ outB,
                       int K, int N) {
  constexpr int MSUB = MT / 16;
  __shared__ __align__(16) __bf16 atile[2][MT * 32];
  int r0 = blockIdx.x * MT;
  int wave = (int)(threadIdx.x >> 5);
  int c0 = blockIdx.y * 256 + wave * 64;
  int l = laneid();
  int n = l & 15;
  int mb = (l < 16) ? 0 : 8;
  int tid = (int)threadIdx.x;

  auto stageA = [&](int buf, int k0) {
#if HAVE_TDM
    if (wave == 0)
      tdm_load_tile(A + (size_t)r0 * K + k0, &atile[buf][0], 32, MT, K);
#else
    for (int t = tid; t < MT * 4; t += 128) {
      int row = t >> 2, seg = (t & 3) * 8;
      async_copy_b128(A + (size_t)(r0 + row) * K + k0 + seg,
                      &atile[buf][row * 32 + seg]);
    }
#endif
  };
  auto waitStage = [&]() {
#if HAVE_TDM
    wait_tensor();
#else
    wait_async();
#endif
  };

  v8f acc[MSUB][4];
#pragma unroll
  for (int i = 0; i < MSUB; ++i)
#pragma unroll
    for (int j = 0; j < 4; ++j) acc[i][j] = zero8();

  stageA(0, 0);
  waitStage();
  __syncthreads();
  int cur = 0;

  for (int k0 = 0; k0 < K; k0 += 32) {
    if (k0 + 32 < K) {
      stageA(cur ^ 1, k0 + 32);
      __builtin_prefetch(Wt + (size_t)(c0 + l) * K + k0 + 32, 0, 1);
    }
    v16bf a[MSUB];
#pragma unroll
    for (int i = 0; i < MSUB; ++i) a[i] = ldA(atile[cur] + i * 16 * 32, 32);
#pragma unroll
    for (int j = 0; j < 4; ++j) {
      v16bf bfr = ldBt(Wt + (size_t)(c0 + 16 * j) * K + k0, K);
#pragma unroll
      for (int i = 0; i < MSUB; ++i) acc[i][j] = wmma_bf16(a[i], bfr, acc[i][j]);
    }
    waitStage();
    __syncthreads();
    cur ^= 1;
  }

#pragma unroll
  for (int j = 0; j < 4; ++j) {
    float bb = bias[c0 + 16 * j + n];
#pragma unroll
    for (int i = 0; i < MSUB; ++i)
#pragma unroll
      for (int r = 0; r < 8; ++r) {
        float v = acc[i][j][r] + bb;
        if (RELU) v = fmaxf(v, 0.f);
        acc[i][j][r] = v;
      }
  }

  if constexpr (DECOMP) {
    __shared__ float tile[MT * 256];
#pragma unroll
    for (int j = 0; j < 4; ++j) {
      int c = c0 + 16 * j + n;  // gridDim.y == 1 here, c < 256
#pragma unroll
      for (int i = 0; i < MSUB; ++i)
#pragma unroll
        for (int r = 0; r < 8; ++r) {
          int row = i * 16 + mb + r;
          float v = acc[i][j][r];
          if (resid) v += resid[(size_t)(r0 + row) * 256 + c];
          tile[row * 256 + c] = v;
        }
    }
    __syncthreads();
    // sliding-window 25-tap moving average (replicate padding via clamping)
    for (int t = tid; t < MT * 8; t += 128) {
      int row = t >> 3, cb = (t & 7) * 32;
      const float* tr = tile + row * 256;
      float run = 0.f;
#pragma unroll
      for (int j = -PADW; j <= PADW; ++j) {
        int cc = cb + j;
        cc = cc < 0 ? 0 : (cc > 255 ? 255 : cc);
        run += tr[cc];
      }
      for (int c = cb; c < cb + 32; ++c) {
        float val = tr[c] - run * (1.f / 25.f);
        size_t o = (size_t)(r0 + row) * 256 + c;
        if (OF) outF[o] = val;
        if (OB) outB[o] = (__bf16)val;
        int cin = c + 13;  if (cin > 255) cin = 255;
        int cout = c - 12; if (cout < 0)  cout = 0;
        run += tr[cin] - tr[cout];
      }
    }
  } else {
#pragma unroll
    for (int j = 0; j < 4; ++j) {
#pragma unroll
      for (int i = 0; i < MSUB; ++i) {
        size_t base = (size_t)(r0 + i * 16 + mb) * N + c0 + 16 * j + n;
#pragma unroll
        for (int r = 0; r < 8; ++r) {
          if (OB) outB[base + (size_t)r * N] = (__bf16)acc[i][j][r];
          if (OF) outF[base + (size_t)r * N] = acc[i][j][r];
        }
      }
    }
  }
}

// ---------------------------------------------------------------------------
extern "C" void kernel_launch(void* const* d_in, const int* in_sizes, int n_in,
                              void* d_out, int out_size, void* d_ws, size_t ws_size,
                              hipStream_t stream) {
  (void)in_sizes; (void)n_in; (void)out_size; (void)ws_size;
  const float* x     = (const float*)d_in[0];
  const int*   mask  = (const int*)d_in[1];
  const float* wq    = (const float*)d_in[2];
  const float* wk    = (const float*)d_in[3];
  const float* wv    = (const float*)d_in[4];
  const float* w_out = (const float*)d_in[5];
  const float* b_out = (const float*)d_in[6];
  const float* ff_w1 = (const float*)d_in[7];
  const float* ff_b1 = (const float*)d_in[8];
  const float* ff_w2 = (const float*)d_in[9];
  const float* ff_b2 = (const float*)d_in[10];
  const float* pr_w1 = (const float*)d_in[11];
  const float* pr_b1 = (const float*)d_in[12];
  const float* pr_w2 = (const float*)d_in[13];
  const float* pr_b2 = (const float*)d_in[14];

  uint8_t* ws = (uint8_t*)d_ws;
  size_t off = 0;
  auto alloc = [&](size_t bytes) -> void* {
    void* p = (void*)(ws + off);
    off += (bytes + 255) & ~(size_t)255;
    return p;
  };
  __bf16* xb    = (__bf16*)alloc((size_t)T_ * E_ * 2);
  __bf16* wqTt  = (__bf16*)alloc((size_t)D_ * D_ * 2);
  __bf16* wkTt  = (__bf16*)alloc((size_t)D_ * D_ * 2);
  __bf16* wvTt  = (__bf16*)alloc((size_t)D_ * D_ * 2);
  __bf16* woutT = (__bf16*)alloc((size_t)E_ * E_ * 2);
  __bf16* ff1T  = (__bf16*)alloc((size_t)E_ * FF_ * 2);
  __bf16* ff2T  = (__bf16*)alloc((size_t)FF_ * E_ * 2);
  __bf16* pr1T  = (__bf16*)alloc((size_t)E_ * FF_ * 2);
  __bf16* pr2T  = (__bf16*)alloc((size_t)FF_ * E_ * 2);
  __bf16* qT    = (__bf16*)alloc((size_t)T_ * E_ * 2);   // [B*H][S][D]
  __bf16* kT    = (__bf16*)alloc((size_t)T_ * E_ * 2);   // [B*H][S][D]
  __bf16* vTT   = (__bf16*)alloc((size_t)T_ * E_ * 2);   // [B*H][D][S]
  __bf16* aob   = (__bf16*)alloc((size_t)T_ * E_ * 2);   // attention out (bf16)
  float*  yF    = (float*) alloc((size_t)T_ * E_ * 4);   // seasonal #1 (f32)
  __bf16* yB    = (__bf16*)alloc((size_t)T_ * E_ * 2);
  __bf16* h1    = (__bf16*)alloc((size_t)T_ * FF_ * 2);
  __bf16* sB    = (__bf16*)alloc((size_t)T_ * E_ * 2);   // seasonal #2 (bf16)
  __bf16* h2    = (__bf16*)alloc((size_t)T_ * FF_ * 2);

  auto cvt = [&](const float* s, __bf16* d, int nElems) {
    k_cvt_bf16<<<(nElems + 255) / 256, 256, 0, stream>>>(s, d, nElems);
  };
  auto cvtT = [&](const float* s, __bf16* d, int R, int C) {
    k_cvt_bf16_T<<<(R * C + 255) / 256, 256, 0, stream>>>(s, d, R, C);
  };
  cvt(x, xb, T_ * E_);
  cvtT(wq, wqTt, D_, D_);
  cvtT(wk, wkTt, D_, D_);
  cvtT(wv, wvTt, D_, D_);
  cvtT(w_out, woutT, E_, E_);
  cvtT(ff_w1, ff1T, E_, FF_);
  cvtT(ff_w2, ff2T, FF_, E_);
  cvtT(pr_w1, pr1T, E_, FF_);
  cvtT(pr_w2, pr2T, FF_, E_);

  // QKV projection: grid (B*H, S/64), 4 waves per block
  k_qkv<<<dim3(B_ * H_, S_ / 64), 128, 0, stream>>>(xb, wqTt, wkTt, wvTt, qT, kT, vTT);

  // Flash attention: grid (B*H, S/64)
  k_attn<<<dim3(B_ * H_, S_ / 64), 128, 0, stream>>>(qT, kT, vTT, mask, aob);

  // attn_out @ w_out + b_out + x, then decomposition -> y   (MT=32, f32+bf16 out)
  k_gemm<32, false, true, true, true><<<dim3(T_ / 32, 1), 128, 0, stream>>>(
      aob, woutT, b_out, x, yF, yB, E_, E_);

  // h1 = relu(y @ ff_w1 + ff_b1)                            (MT=64, bf16 out)
  k_gemm<64, true, false, false, true><<<dim3(T_ / 64, FF_ / 256), 128, 0, stream>>>(
      yB, ff1T, ff_b1, nullptr, nullptr, h1, E_, FF_);

  // s = decomp(y + h1 @ ff_w2 + ff_b2)                      (MT=32, bf16 out)
  k_gemm<32, false, true, false, true><<<dim3(T_ / 32, 1), 128, 0, stream>>>(
      h1, ff2T, ff_b2, yF, nullptr, sB, FF_, E_);

  // h2 = relu(s @ pr_w1 + pr_b1)                            (MT=64, bf16 out)
  k_gemm<64, true, false, false, true><<<dim3(T_ / 64, FF_ / 256), 128, 0, stream>>>(
      sB, pr1T, pr_b1, nullptr, nullptr, h2, E_, FF_);

  // out = h2 @ pr_w2 + pr_b2                                (MT=64, f32 out)
  k_gemm<64, false, false, true, false><<<dim3(T_ / 64, 1), 128, 0, stream>>>(
      h2, pr2T, pr_b2, nullptr, (float*)d_out, nullptr, FF_, E_);
}
